// MultiHeadAttention_10333691314638
// MI455X (gfx1250) — matrix operand order repaired and checked by reading, not compiled, not run on hardware
//
#include <hip/hip_runtime.h>

// ---------------------------------------------------------------------------
// MHA forward for MI455X (gfx1250, wave32, WMMA 16x16x32 bf16 / f32 accum)
//   B=2, S=2048, D=1024, H=16, DQK=DH=64
// ---------------------------------------------------------------------------

typedef __bf16 bf16;
typedef __attribute__((ext_vector_type(16))) __bf16 v16bf;
typedef __attribute__((ext_vector_type(8)))  __bf16 v8bf;
typedef __attribute__((ext_vector_type(8)))  float  v8f;

#define DEV __device__ __forceinline__

static constexpr int Bc = 2;
static constexpr int Sc = 2048;
static constexpr int Dc = 1024;
static constexpr int Hc = 16;
static constexpr int Ec = 64;   // head dim

DEV v8f vzero8() {
  v8f z;
#pragma unroll
  for (int i = 0; i < 8; ++i) z[i] = 0.0f;
  return z;
}

DEV v16bf concat8(v8bf lo, v8bf hi) {
  return __builtin_shufflevector(lo, hi, 0, 1, 2, 3, 4, 5, 6, 7,
                                 8, 9, 10, 11, 12, 13, 14, 15);
}

DEV v8f wmma_bf16(v16bf a, v16bf b, v8f c) {
  // (neg_a, A, neg_b, B, c_mod, C, reuse_a, reuse_b)
  return __builtin_amdgcn_wmma_f32_16x16x32_bf16(false, a, false, b,
                                                 (short)0, c, false, false);
}

// --- A fragment (16x32, 16-bit): lane holds row M=lane&15,
//     element e -> K = (e&7) + ((e>>3)<<4) + 8*(lane>=16)
DEV v16bf load_a_f32(const float* __restrict__ base, int ld, int k0) {
  int lane = threadIdx.x & 31;
  const float* p = base + (size_t)(lane & 15) * ld + k0 + ((lane >> 4) << 3);
  v8f f0 = *(const v8f*)p;          // K run  k0+off .. +7   (2x b128)
  v8f f1 = *(const v8f*)(p + 16);   // K run  k0+off+16 .. +7
  return concat8(__builtin_convertvector(f0, v8bf),
                 __builtin_convertvector(f1, v8bf));
}

DEV v16bf load_a_bf16(const bf16* __restrict__ base, int ld, int k0) {
  int lane = threadIdx.x & 31;
  const bf16* p = base + (size_t)(lane & 15) * ld + k0 + ((lane >> 4) << 3);
  v8bf lo = *(const v8bf*)p;          // 16B load
  v8bf hv = *(const v8bf*)(p + 16);   // 16B load
  return concat8(lo, hv);
}

// --- B fragment (32x16, 16-bit): lane holds col N=lane&15,
//     element e -> K = (e&7) + ((e>>3)<<4) + 8*(lane>=16).
// Source must be bf16 with the K-run contiguous (ld = stride between columns).
DEV v16bf load_b_contig(const bf16* __restrict__ base, int ld, int k0) {
  int lane = threadIdx.x & 31;
  const bf16* p = base + (size_t)(lane & 15) * ld + k0 + ((lane >> 4) << 3);
  v8bf x0 = *(const v8bf*)p;          // 16B load
  v8bf x1 = *(const v8bf*)(p + 16);   // 16B load
  return concat8(x0, x1);
}

// ---------------------------------------------------------------------------
// Kernel 0: one-time weight prep.  Converts f32 weights to bf16 laid out
// [N, K] row-major so B-fragment loads are contiguous 16B per lane.
//   blockIdx.y: 0=Wq 1=Wk 2=Wv (each [H,D,64] -> [H*64, D])
//               3=Wo ([D,D] row-major k,n -> [n, k])
// grid = (1024*1024/256, 4)
// ---------------------------------------------------------------------------
__global__ __launch_bounds__(256) void prep_weights_kernel(
    const float* __restrict__ Wq, const float* __restrict__ Wk,
    const float* __restrict__ Wv, const float* __restrict__ Wo,
    bf16* __restrict__ WqT, bf16* __restrict__ WkT,
    bf16* __restrict__ WvT, bf16* __restrict__ WoT) {
  const float* src; bf16* dst; bool qkv;
  if (blockIdx.y == 0)      { src = Wq; dst = WqT; qkv = true; }
  else if (blockIdx.y == 1) { src = Wk; dst = WkT; qkv = true; }
  else if (blockIdx.y == 2) { src = Wv; dst = WvT; qkv = true; }
  else                      { src = Wo; dst = WoT; qkv = false; }

  int i = blockIdx.x * 256 + threadIdx.x;   // coalesced read index
  float v = src[i];
  int n, k;
  if (qkv) {                                // i = h*D*64 + d*64 + e
    int h = i >> 16, d = (i >> 6) & (Dc - 1), e = i & 63;
    n = h * Ec + e; k = d;
  } else {                                  // i = k*D + n
    k = i >> 10; n = i & (Dc - 1);
  }
  dst[(size_t)n * Dc + k] = (bf16)v;
}

// ---------------------------------------------------------------------------
// Kernel 1: QKV projection.  grid=(M/256, 1024/64, 3), block=256 (8 waves).
// Each wave: 32 rows x 64 cols (2 A-tiles x 4 B-tiles, B reused across A),
// K-loop over D in steps of 32, software-pipelined (double-buffered).
// Q,K -> bf16 [B*H, S, 64];  V -> transposed bf16 [B*H, 64, S].
// ---------------------------------------------------------------------------
__global__ __launch_bounds__(256) void qkv_proj_kernel(
    const float* __restrict__ x,
    const bf16* __restrict__ WqT, const float* __restrict__ bq,
    const bf16* __restrict__ WkT, const float* __restrict__ bk,
    const bf16* __restrict__ WvT, const float* __restrict__ bv,
    bf16* __restrict__ Q, bf16* __restrict__ K, bf16* __restrict__ VT) {
  const bf16* W; const float* bias; bf16* dst;
  bool transV = false;
  if (blockIdx.z == 0)      { W = WqT; bias = bq; dst = Q; }
  else if (blockIdx.z == 1) { W = WkT; bias = bk; dst = K; }
  else                      { W = WvT; bias = bv; dst = VT; transV = true; }

  int w = threadIdx.x >> 5, lane = threadIdx.x & 31;
  int lo16 = lane & 15, hi = lane >> 4;
  int m0 = blockIdx.x * 256 + w * 32;
  int n0 = blockIdx.y * 64;

  v8f acc[2][4];
#pragma unroll
  for (int rt = 0; rt < 2; ++rt)
#pragma unroll
    for (int j = 0; j < 4; ++j) acc[rt][j] = vzero8();

  const float* arow0 = x + (size_t)m0 * Dc;
  const float* arow1 = arow0 + (size_t)16 * Dc;

  // pipeline prologue
  v16bf a0 = load_a_f32(arow0, Dc, 0);
  v16bf a1 = load_a_f32(arow1, Dc, 0);
  v16bf b[4];
#pragma unroll
  for (int j = 0; j < 4; ++j)
    b[j] = load_b_contig(W + (size_t)(n0 + j * 16) * Dc, Dc, 0);

  for (int k0 = 0; k0 < Dc - 32; k0 += 32) {
    int kn = k0 + 32;
    v16bf a0n = load_a_f32(arow0, Dc, kn);
    v16bf a1n = load_a_f32(arow1, Dc, kn);
    v16bf bn[4];
#pragma unroll
    for (int j = 0; j < 4; ++j)
      bn[j] = load_b_contig(W + (size_t)(n0 + j * 16) * Dc, Dc, kn);
#pragma unroll
    for (int j = 0; j < 4; ++j) {
      acc[0][j] = wmma_bf16(a0, b[j], acc[0][j]);
      acc[1][j] = wmma_bf16(a1, b[j], acc[1][j]);
    }
    a0 = a0n; a1 = a1n;
#pragma unroll
    for (int j = 0; j < 4; ++j) b[j] = bn[j];
  }
#pragma unroll
  for (int j = 0; j < 4; ++j) {           // pipeline epilogue (last k-step)
    acc[0][j] = wmma_bf16(a0, b[j], acc[0][j]);
    acc[1][j] = wmma_bf16(a1, b[j], acc[1][j]);
  }

#pragma unroll
  for (int rt = 0; rt < 2; ++rt) {
#pragma unroll
    for (int j = 0; j < 4; ++j) {
      int n = n0 + j * 16 + lo16;
      float bb = bias[n];
      int h = n >> 6, e = n & 63;
#pragma unroll
      for (int r = 0; r < 8; ++r) {
        int m = m0 + rt * 16 + r + 8 * hi;       // global row = b*S + s
        int bidx = m >> 11, s = m & (Sc - 1);
        float val = acc[rt][j][r] + bb;
        size_t bhOff = (size_t)(bidx * Hc + h);
        if (transV)
          dst[bhOff * Ec * Sc + (size_t)e * Sc + s] = (bf16)val;     // [64,S]
        else
          dst[(bhOff * Sc + s) * Ec + e] = (bf16)val;                // [S,64]
      }
    }
  }
}

// ---------------------------------------------------------------------------
// Kernel 2: causal flash attention.  grid=(S/128, B*H), block=256 (8 waves).
// Each wave owns 16 query rows; streams 32-key tiles with online softmax.
// V fragments are fetched before softmax so their latency hides under VALU.
// P (C-layout) is re-staged through LDS into A-fragment layout for P@V.
// ---------------------------------------------------------------------------
__global__ __launch_bounds__(256) void flash_attn_kernel(
    const bf16* __restrict__ Q, const bf16* __restrict__ K,
    const bf16* __restrict__ VT, bf16* __restrict__ O) {
  __shared__ __align__(16) bf16 lds_p[8 * 16 * 32];   // 8 KB, one 16x32 tile/wave

  int w = threadIdx.x >> 5, lane = threadIdx.x & 31;
  int lo16 = lane & 15, hi = lane >> 4;
  int bh = blockIdx.y;
  const bf16* Qb  = Q  + (size_t)bh * Sc * Ec;
  const bf16* Kb  = K  + (size_t)bh * Sc * Ec;
  const bf16* VTb = VT + (size_t)bh * Ec * Sc;        // [64, S]
  int q0 = blockIdx.x * 128 + w * 16;

  v16bf qa0 = load_a_bf16(Qb + (size_t)q0 * Ec, Ec, 0);
  v16bf qa1 = load_a_bf16(Qb + (size_t)q0 * Ec, Ec, 32);

  float mrow[8], lrow[8];
  v8f acc[4];
#pragma unroll
  for (int r = 0; r < 8; ++r) { mrow[r] = -1e30f; lrow[r] = 0.0f; }
#pragma unroll
  for (int j = 0; j < 4; ++j) acc[j] = vzero8();

  bf16* lp = lds_p + w * 512;

  for (int t0 = 0; t0 < q0 + 16; t0 += 32) {
    // prefetch next key tile while this one computes
    __builtin_prefetch((const void*)(Kb + (size_t)(t0 + 32) * Ec), 0, 1);

    // --- S = (Q K^T) * scale, two 16x16 tiles covering 32 keys
    v16bf kb00 = load_b_contig(Kb + (size_t)t0 * Ec, Ec, 0);
    v16bf kb01 = load_b_contig(Kb + (size_t)t0 * Ec, Ec, 32);
    v16bf kb10 = load_b_contig(Kb + (size_t)(t0 + 16) * Ec, Ec, 0);
    v16bf kb11 = load_b_contig(Kb + (size_t)(t0 + 16) * Ec, Ec, 32);
    v8f s0 = vzero8(), s1 = vzero8();
    s0 = wmma_bf16(qa0, kb00, s0);
    s0 = wmma_bf16(qa1, kb01, s0);
    s1 = wmma_bf16(qa0, kb10, s1);
    s1 = wmma_bf16(qa1, kb11, s1);

    // --- issue V fragments now; independent of scores, hides under softmax
    v16bf vb[4];
#pragma unroll
    for (int j = 0; j < 4; ++j)
      vb[j] = load_b_contig(VTb + (size_t)(j * 16) * Sc + t0, Sc, 0);

    int tA = t0 + lo16, tB = t0 + 16 + lo16;
#pragma unroll
    for (int r = 0; r < 8; ++r) {
      int qrow = q0 + r + 8 * hi;
      float a0 = s0[r] * 0.125f;                 // 1/sqrt(64)
      float a1 = s1[r] * 0.125f;
      if (tA > qrow) a0 = -1e30f;                // causal mask
      if (tB > qrow) a1 = -1e30f;

      // row max across the 16-lane half (row lives across lanes 0..15 / 16..31)
      float rmx = fmaxf(a0, a1);
      rmx = fmaxf(rmx, __shfl_xor(rmx, 1, 32));
      rmx = fmaxf(rmx, __shfl_xor(rmx, 2, 32));
      rmx = fmaxf(rmx, __shfl_xor(rmx, 4, 32));
      rmx = fmaxf(rmx, __shfl_xor(rmx, 8, 32));

      float mnew  = fmaxf(mrow[r], rmx);
      float alpha = __expf(mrow[r] - mnew);
      float p0 = __expf(a0 - mnew);
      float p1 = __expf(a1 - mnew);
      float rs = p0 + p1;
      rs += __shfl_xor(rs, 1, 32);
      rs += __shfl_xor(rs, 2, 32);
      rs += __shfl_xor(rs, 4, 32);
      rs += __shfl_xor(rs, 8, 32);

      lrow[r] = lrow[r] * alpha + rs;
      mrow[r] = mnew;
#pragma unroll
      for (int j = 0; j < 4; ++j) acc[j][r] *= alpha;
      s0[r] = p0; s1[r] = p1;
    }

    // --- stage P (C-layout) into LDS as row-major 16x32 bf16
    asm volatile("" ::: "memory");
#pragma unroll
    for (int r = 0; r < 8; ++r) {
      int row = r + 8 * hi;
      lp[row * 32 + lo16]      = (bf16)s0[r];
      lp[row * 32 + 16 + lo16] = (bf16)s1[r];
    }
    asm volatile("s_wait_dscnt 0" ::: "memory");   // LDS RAW: stores -> A-frag load

    v16bf pa = load_a_bf16(lp, 32, 0);

    // --- O += P @ V  (V fragments already in registers)
#pragma unroll
    for (int j = 0; j < 4; ++j)
      acc[j] = wmma_bf16(pa, vb[j], acc[j]);
  }

  // --- normalize and store bf16 activation [B*S, H*64]
  int b = bh >> 4, h = bh & 15;
#pragma unroll
  for (int r = 0; r < 8; ++r) {
    float inv = 1.0f / lrow[r];
    int s = q0 + r + 8 * hi;
    size_t rowoff = ((size_t)b * Sc + s) * (size_t)Dc + (size_t)h * Ec;
#pragma unroll
    for (int j = 0; j < 4; ++j)
      O[rowoff + j * 16 + lo16] = (bf16)(acc[j][r] * inv);
  }
}

// ---------------------------------------------------------------------------
// Kernel 3: output projection.  grid=(M/256, 1024/64), block=256.
// out[4096,1024] (f32) = A(bf16)[4096,1024] @ WoT(bf16 [n,k]) + bo
// Same 32x64-per-wave, double-buffered scheme as kernel 1.
// ---------------------------------------------------------------------------
__global__ __launch_bounds__(256) void out_proj_kernel(
    const bf16* __restrict__ A, const bf16* __restrict__ WoT,
    const float* __restrict__ bo, float* __restrict__ out) {
  int w = threadIdx.x >> 5, lane = threadIdx.x & 31;
  int lo16 = lane & 15, hi = lane >> 4;
  int m0 = blockIdx.x * 256 + w * 32;
  int n0 = blockIdx.y * 64;

  v8f acc[2][4];
#pragma unroll
  for (int rt = 0; rt < 2; ++rt)
#pragma unroll
    for (int j = 0; j < 4; ++j) acc[rt][j] = vzero8();

  const bf16* arow0 = A + (size_t)m0 * Dc;
  const bf16* arow1 = arow0 + (size_t)16 * Dc;

  v16bf a0 = load_a_bf16(arow0, Dc, 0);
  v16bf a1 = load_a_bf16(arow1, Dc, 0);
  v16bf b[4];
#pragma unroll
  for (int j = 0; j < 4; ++j)
    b[j] = load_b_contig(WoT + (size_t)(n0 + j * 16) * Dc, Dc, 0);

  for (int k0 = 0; k0 < Dc - 32; k0 += 32) {
    int kn = k0 + 32;
    v16bf a0n = load_a_bf16(arow0, Dc, kn);
    v16bf a1n = load_a_bf16(arow1, Dc, kn);
    v16bf bn[4];
#pragma unroll
    for (int j = 0; j < 4; ++j)
      bn[j] = load_b_contig(WoT + (size_t)(n0 + j * 16) * Dc, Dc, kn);
#pragma unroll
    for (int j = 0; j < 4; ++j) {
      acc[0][j] = wmma_bf16(a0, b[j], acc[0][j]);
      acc[1][j] = wmma_bf16(a1, b[j], acc[1][j]);
    }
    a0 = a0n; a1 = a1n;
#pragma unroll
    for (int j = 0; j < 4; ++j) b[j] = bn[j];
  }
#pragma unroll
  for (int j = 0; j < 4; ++j) {
    acc[0][j] = wmma_bf16(a0, b[j], acc[0][j]);
    acc[1][j] = wmma_bf16(a1, b[j], acc[1][j]);
  }

#pragma unroll
  for (int rt = 0; rt < 2; ++rt) {
#pragma unroll
    for (int j = 0; j < 4; ++j) {
      int n = n0 + j * 16 + lo16;
      float bb = bo[n];
#pragma unroll
      for (int r = 0; r < 8; ++r) {
        int m = m0 + rt * 16 + r + 8 * hi;
        out[(size_t)m * Dc + n] = acc[rt][j][r] + bb;
      }
    }
  }
}

// ---------------------------------------------------------------------------
extern "C" void kernel_launch(void* const* d_in, const int* in_sizes, int n_in,
                              void* d_out, int out_size, void* d_ws, size_t ws_size,
                              hipStream_t stream) {
  const float* x  = (const float*)d_in[0];
  const float* Wq = (const float*)d_in[1];
  const float* bq = (const float*)d_in[2];
  const float* Wk = (const float*)d_in[3];
  const float* bk = (const float*)d_in[4];
  const float* Wv = (const float*)d_in[5];
  const float* bv = (const float*)d_in[6];
  const float* Wo = (const float*)d_in[7];
  const float* bo = (const float*)d_in[8];
  float* out = (float*)d_out;

  const size_t seg  = (size_t)Bc * Hc * Sc * Ec;  // 4,194,304 elems per tensor
  const size_t wseg = (size_t)Dc * Dc;            // 1,048,576 elems per weight
  bf16* Q   = (bf16*)d_ws;
  bf16* K   = Q + seg;
  bf16* VT  = K + seg;        // [B*H, 64, S]
  bf16* A   = VT + seg;       // [B*S, H*64]
  bf16* WqT = A + seg;
  bf16* WkT = WqT + wseg;
  bf16* WvT = WkT + wseg;
  bf16* WoT = WvT + wseg;

  dim3 blk(256);
  prep_weights_kernel<<<dim3(wseg / 256, 4), blk, 0, stream>>>(
      Wq, Wk, Wv, Wo, WqT, WkT, WvT, WoT);
  qkv_proj_kernel<<<dim3((Bc * Sc) / 256, Dc / 64, 3), blk, 0, stream>>>(
      x, WqT, bq, WkT, bk, WvT, bv, Q, K, VT);
  flash_attn_kernel<<<dim3(Sc / 128, Bc * Hc), blk, 0, stream>>>(Q, K, VT, A);
  out_proj_kernel<<<dim3((Bc * Sc) / 256, Dc / 64), blk, 0, stream>>>(A, WoT, bo, out);
}
